// AirFormer_1726576853793
// MI455X (gfx1250) — matrix operand order repaired and checked by reading, not compiled, not run on hardware
//
#include <hip/hip_runtime.h>
#include <hip/hip_bf16.h>
#include <math.h>

// ---------------------------------------------------------------------------
// AirFormer on MI455X (gfx1250).
// Compute-bound (~190 GFLOP vs ~25MB activations @23.3TB/s) -> all big GEMMs
// on v_wmma_f32_16x16x32_f16 (f32 accum). GEMM operands live in memory as f16
// (weights converted+transposed once, activations mirrored by epilogues), so
// LDS tiles can be filled with async global->LDS B128 copies (ASYNCcnt path)
// when the toolchain exposes the builtins; otherwise vectorized v8h copies.
// ---------------------------------------------------------------------------

typedef _Float16 v16h __attribute__((ext_vector_type(16)));
typedef _Float16 v8h  __attribute__((ext_vector_type(8)));
typedef float    v8f  __attribute__((ext_vector_type(8)));
typedef int      b128_t __attribute__((vector_size(16)));  // matches builtin V4i

#define GLOBAL_AS __attribute__((address_space(1)))
#define LOCAL_AS  __attribute__((address_space(3)))

#if defined(__has_builtin)
#if __has_builtin(__builtin_amdgcn_global_load_async_to_lds_b128) && \
    __has_builtin(__builtin_amdgcn_s_wait_asynccnt)
#define HAVE_ASYNC_LDS 1
#endif
#endif
#ifndef HAVE_ASYNC_LDS
#define HAVE_ASYNC_LDS 0
#endif

// Copy 16B (8 halves) global->LDS, async (ASYNCcnt path) when available.
__device__ __forceinline__ void copy16_g2l(_Float16* lds_dst, const _Float16* g_src) {
#if HAVE_ASYNC_LDS
  __builtin_amdgcn_global_load_async_to_lds_b128(
      (GLOBAL_AS b128_t*)g_src, (LOCAL_AS b128_t*)lds_dst, 0, 0);
#else
  *(v8h*)lds_dst = *(const v8h*)g_src;
#endif
}
__device__ __forceinline__ void wait_async_copies() {
#if HAVE_ASYNC_LDS
  __builtin_amdgcn_s_wait_asynccnt(0);
#endif
}

// A-fragment (16x32 f16, ISA 7.12.2): lane<16 holds K {0..7,16..23}, lane>=16
// holds K {8..15,24..31}; two 16B LDS loads at p and p+16.
__device__ __forceinline__ v16h frag_a(const _Float16* p) {
  union { v16h v; v8h h[2]; } u;
  u.h[0] = *(const v8h*)(p);
  u.h[1] = *(const v8h*)(p + 16);
  return u.v;
}
// B-fragment (32x16 f16): same K order as A-fragment; two 16B loads at p and p+16.
__device__ __forceinline__ v16h frag_b(const _Float16* p) {
  union { v16h v; v8h h[2]; } u;
  u.h[0] = *(const v8h*)(p);
  u.h[1] = *(const v8h*)(p + 16);
  return u.v;
}

// ---------------------------------------------------------------------------
// Generic WMMA GEMM: C[M,Nn] = epilogue(A[M,K] @ B[K,Nn]) with A f16 row-major
// and B supplied TRANSPOSED f16 (Bt[n*K+k]). Block tile 128x64, BK=32,
// 256 threads (8 wave32s, each 32x32 = 2x2 WMMA tiles).
// Requires M%128==0, Nn%64==0, K%32==0 (true for all uses here).
// ---------------------------------------------------------------------------
template <int HAS_BIAS, int DO_GELU, int HAS_RES, int STORE_F16>
__global__ __launch_bounds__(256)
void gemm_wmma(const _Float16* __restrict__ A, const _Float16* __restrict__ Bt,
               const float* __restrict__ bias, const float* __restrict__ res,
               float* __restrict__ C, _Float16* __restrict__ C16,
               int M, int K, int Nn) {
  __shared__ _Float16 sA[128 * 40];  // K padded 32->40 halves (16B-aligned rows)
  __shared__ _Float16 sB[64 * 40];   // sB[n*40 + k]

  const int tid  = threadIdx.x;
  const int wave = tid >> 5;
  const int lane = tid & 31;
  const int lr   = lane & 15;
  const int kbA  = (lane & 16) ? 8 : 0;   // A-frag K base
  const int kbB  = (lane & 16) ? 8 : 0;   // B-frag K base (0 or 8)
  const int rsel = (lane & 16) ? 8 : 0;   // C/D row select
  const int bm   = blockIdx.y * 128;
  const int bn   = blockIdx.x * 64;
  const int wm   = (wave >> 1) * 32;
  const int wn   = (wave & 1) * 32;

  v8f acc[2][2];
#pragma unroll
  for (int i = 0; i < 2; i++)
#pragma unroll
    for (int j = 0; j < 2; j++)
#pragma unroll
      for (int e = 0; e < 8; e++) acc[i][j][e] = 0.0f;

  for (int kt = 0; kt < K; kt += 32) {
    // Stage A tile 128x32 halves: 512 x 16B segments, 2 per thread.
#pragma unroll
    for (int i = 0; i < 2; i++) {
      int idx = tid + i * 256;
      int r = idx >> 2, k8 = (idx & 3) * 8;
      copy16_g2l(&sA[r * 40 + k8], &A[(size_t)(bm + r) * K + kt + k8]);
    }
    // Stage B tile 64x32 halves: 256 x 16B segments, 1 per thread.
    {
      int r = tid >> 2, k8 = (tid & 3) * 8;
      copy16_g2l(&sB[r * 40 + k8], &Bt[(size_t)(bn + r) * K + kt + k8]);
    }
    // Prefetch next K tile of A into caches.
    if (kt + 32 < K) {
      int r = tid >> 2, k8 = (tid & 3) * 8;
      __builtin_prefetch(&A[(size_t)(bm + r) * K + kt + 32 + k8], 0, 1);
    }
    wait_async_copies();
    __syncthreads();

    v16h af[2], bf[2];
#pragma unroll
    for (int mt = 0; mt < 2; mt++)
      af[mt] = frag_a(&sA[(wm + mt * 16 + lr) * 40 + kbA]);
#pragma unroll
    for (int nt = 0; nt < 2; nt++)
      bf[nt] = frag_b(&sB[(wn + nt * 16 + lr) * 40 + kbB]);

#pragma unroll
    for (int mt = 0; mt < 2; mt++)
#pragma unroll
      for (int nt = 0; nt < 2; nt++)
        acc[mt][nt] = __builtin_amdgcn_wmma_f32_16x16x32_f16(
            false, af[mt], false, bf[nt], (short)0, acc[mt][nt], false, false);
    __syncthreads();
  }

  // Epilogue
#pragma unroll
  for (int mt = 0; mt < 2; mt++) {
#pragma unroll
    for (int nt = 0; nt < 2; nt++) {
      const int col  = bn + wn + nt * 16 + lr;
      const int row0 = bm + wm + mt * 16 + rsel;
#pragma unroll
      for (int v = 0; v < 8; v++) {
        const int row = row0 + v;
        float val = acc[mt][nt][v];
        if constexpr (HAS_BIAS) val += bias[col];
        if constexpr (DO_GELU) val = 0.5f * val * (1.0f + erff(val * 0.70710678118f));
        if constexpr (HAS_RES) val += res[(size_t)row * Nn + col];
        C[(size_t)row * Nn + col] = val;
        if constexpr (STORE_F16) C16[(size_t)row * Nn + col] = (_Float16)val;
      }
    }
  }
}

// ---------------------------------------------------------------------------
// Weight convert+transpose: W[K,Nn] f32 -> WT[Nn,K] f16
// ---------------------------------------------------------------------------
__global__ __launch_bounds__(256)
void convertT(const float* __restrict__ W, _Float16* __restrict__ WT, int K, int Nn) {
  int idx = blockIdx.x * 256 + threadIdx.x;
  if (idx >= K * Nn) return;
  int k = idx / Nn, n = idx % Nn;
  WT[(size_t)n * K + k] = (_Float16)W[idx];
}

// ---------------------------------------------------------------------------
// Embedding / layout shuffles (also produce f16 mirrors for GEMM A inputs)
// ---------------------------------------------------------------------------
__global__ __launch_bounds__(256)
void embed_kernel(const float* __restrict__ x, const float* __restrict__ pos,
                  float* __restrict__ h, _Float16* __restrict__ h16) {
  size_t idx = (size_t)blockIdx.x * 256 + threadIdx.x;  // over 49152*128
  int c = (int)(idx & 127);
  size_t rowi = idx >> 7;          // (b*512+n)*24 + t
  int t = (int)(rowi % 24);
  size_t bn = rowi / 24;
  int nn = (int)(bn & 511);
  int b = (int)(bn >> 9);
  float v = x[(((size_t)b * 128 + c) * 512 + nn) * 24 + t] + pos[t * 128 + c];
  h[idx] = v;
  h16[idx] = (_Float16)v;
}

__global__ __launch_bounds__(256)
void t2s_kernel(const float* __restrict__ h, float* __restrict__ s,
                _Float16* __restrict__ s16) {
  size_t idx = (size_t)blockIdx.x * 256 + threadIdx.x;
  int c = (int)(idx & 127);
  size_t srow = idx >> 7;          // (b*24+t)*512 + n
  int nn = (int)(srow & 511);
  size_t bt = srow >> 9;
  int t = (int)(bt % 24);
  int b = (int)(bt / 24);
  float v = h[(((size_t)b * 512 + nn) * 24 + t) * 128 + c];
  s[idx] = v;
  s16[idx] = (_Float16)v;
}

__global__ __launch_bounds__(256)
void final_kernel(const float* __restrict__ s, float* __restrict__ out) {
  size_t idx = (size_t)blockIdx.x * 256 + threadIdx.x;  // over B*C*N*T
  int t = (int)(idx % 24);
  size_t r1 = idx / 24;
  int nn = (int)(r1 % 512);
  size_t r2 = r1 / 512;
  int c = (int)(r2 % 128);
  int b = (int)(r2 / 128);
  out[idx] = s[(((size_t)b * 24 + t) * 512 + nn) * 128 + c];
}

// ---------------------------------------------------------------------------
// LayerNorm over C=128: one wave32 per row (8 rows/block); writes f32 + f16.
// ---------------------------------------------------------------------------
__global__ __launch_bounds__(256)
void layernorm(const float* __restrict__ X, const float* __restrict__ g,
               const float* __restrict__ b, float* __restrict__ Y,
               _Float16* __restrict__ Y16, int M) {
  const int wave = threadIdx.x >> 5;
  const int lane = threadIdx.x & 31;
  const int row = blockIdx.x * 8 + wave;
  if (row >= M) return;
  const float* x = X + (size_t)row * 128;
  float vals[4], s1 = 0.f, s2 = 0.f;
#pragma unroll
  for (int j = 0; j < 4; j++) {
    float t = x[lane + 32 * j];
    vals[j] = t; s1 += t; s2 += t * t;
  }
#pragma unroll
  for (int off = 16; off; off >>= 1) {
    s1 += __shfl_xor(s1, off);
    s2 += __shfl_xor(s2, off);
  }
  const float mu = s1 * (1.0f / 128.0f);
  const float var = s2 * (1.0f / 128.0f) - mu * mu;
  const float inv = rsqrtf(var + 1e-5f);
  float* y = Y + (size_t)row * 128;
  _Float16* y16 = Y16 + (size_t)row * 128;
#pragma unroll
  for (int j = 0; j < 4; j++) {
    int c = lane + 32 * j;
    float v = (vals[j] - mu) * inv * g[c] + b[c];
    y[c] = v;
    y16[c] = (_Float16)v;
  }
}

// ---------------------------------------------------------------------------
// Temporal windowed causal attention (WS=12, hd=32). qkv: [M,384], column
// layout kk*128 + h*32 + e. One block per (sequence, window, head).
// ---------------------------------------------------------------------------
__global__ __launch_bounds__(256)
void temporal_attn(const float* __restrict__ qkv, float* __restrict__ ctx,
                   _Float16* __restrict__ ctx16) {
  __shared__ float q[12][32], k[12][32], v[12][32];
  __shared__ float p[12][12];
  const int bi = blockIdx.x;
  const int head = bi & 3;
  const int win = (bi >> 2) & 1;
  const int seq = bi >> 3;
  const int rowBase = seq * 24 + win * 12;
  const int tid = threadIdx.x;

  for (int i = tid; i < 384; i += 256) {
    int r = i >> 5, e = i & 31;
    size_t base = (size_t)(rowBase + r) * 384 + head * 32 + e;
    q[r][e] = qkv[base];
    k[r][e] = qkv[base + 128];
    v[r][e] = qkv[base + 256];
  }
  __syncthreads();

  if (tid < 144) {
    int qi = tid / 12, ki = tid % 12;
    float s;
    if (ki > qi) {
      s = -INFINITY;
    } else {
      s = 0.f;
#pragma unroll
      for (int e = 0; e < 32; e++) s += q[qi][e] * k[ki][e];
      s *= 0.17677669529f;  // 1/sqrt(32)
    }
    p[qi][ki] = s;
  }
  __syncthreads();
  if (tid < 12) {
    float m = -INFINITY;
    for (int ki = 0; ki < 12; ki++) m = fmaxf(m, p[tid][ki]);
    float sum = 0.f;
    for (int ki = 0; ki < 12; ki++) {
      float e = expf(p[tid][ki] - m);
      p[tid][ki] = e; sum += e;
    }
    float inv = 1.0f / sum;
    for (int ki = 0; ki < 12; ki++) p[tid][ki] *= inv;
  }
  __syncthreads();
  for (int i = tid; i < 384; i += 256) {
    int qi = i >> 5, e = i & 31;
    float o = 0.f;
#pragma unroll
    for (int ki = 0; ki < 12; ki++) o += p[qi][ki] * v[ki][e];
    size_t oidx = (size_t)(rowBase + qi) * 128 + head * 32 + e;
    ctx[oidx] = o;
    ctx16[oidx] = (_Float16)o;
  }
}

// ---------------------------------------------------------------------------
// sec[m,n] = argmax_r assignment[m,n,r]
// ---------------------------------------------------------------------------
__global__ __launch_bounds__(256)
void compute_sec(const float* __restrict__ assign, int* __restrict__ sec) {
  int idx = blockIdx.x * 256 + threadIdx.x;
  if (idx >= 512 * 512) return;
  const float* a = assign + (size_t)idx * 17;
  int best = 0;
  float bv = a[0];
#pragma unroll
  for (int r = 1; r < 17; r++)
    if (a[r] > bv) { bv = a[r]; best = r; }
  sec[idx] = best;
}

// ---------------------------------------------------------------------------
// Fused spatial attention: per token n,
//   pre_kv[r,c] = sum_{n'} [sec[n,n']==r] * s[bt,n',c]     (LDS f32 atomics)
//   kv[32x256]  = pad(pre_kv) @ kvw                        (WMMA, kvw in LDS)
//   attn over R=17 regions per head, bias + mask, softmax, weighted V-sum.
// kvw arrives pre-transposed f16 (kvwT_g[n*128+k]); staged once per block
// (async B128 when available) and stays resident; 4 tokens per block.
// ---------------------------------------------------------------------------
__global__ __launch_bounds__(256)
void spatial_attn_fused(const float* __restrict__ S, const float* __restrict__ Q,
                        const int* __restrict__ sec,
                        const _Float16* __restrict__ kvwT_g,
                        const float* __restrict__ bias,
                        const unsigned char* __restrict__ mask,
                        float* __restrict__ ctx, _Float16* __restrict__ ctx16) {
  extern __shared__ char smem_raw[];
  _Float16* kvwT = (_Float16*)smem_raw;        // [256][136] (k padded 128->136)
  _Float16* pa   = kvwT + 256 * 136;           // [32][136]
  float* pkv = (float*)(pa + 32 * 136);        // [17][128]
  float* kvb = pkv + 17 * 128;                 // [17][256]
  float* qs  = kvb + 17 * 256;                 // [128]
  float* sc  = qs + 128;                       // [68]
  float* pr  = sc + 68;                        // [68]

  const int tid  = threadIdx.x;
  const int wave = tid >> 5;
  const int lane = tid & 31;
  const int lr   = lane & 15;
  const int kbA  = (lane & 16) ? 8 : 0;
  const int kbB  = (lane & 16) ? 8 : 0;
  const int rsel = (lane & 16) ? 8 : 0;
  const int bt   = blockIdx.y;

  // Stage kvwT (256x128 halves = 4096 x 16B segments, 16 per thread)
#pragma unroll
  for (int i = 0; i < 16; i++) {
    int idx = tid + i * 256;
    int n = idx >> 4, k8 = (idx & 15) * 8;
    copy16_g2l(&kvwT[n * 136 + k8], &kvwT_g[(size_t)n * 128 + k8]);
  }
  wait_async_copies();
  __syncthreads();

  for (int t = 0; t < 4; t++) {
    const int n = blockIdx.x * 4 + t;
    const size_t row = (size_t)bt * 512 + n;

    for (int i = tid; i < 17 * 128; i += 256) pkv[i] = 0.f;
    if (tid < 128) qs[tid] = Q[row * 128 + tid];
    __syncthreads();

    // Segment-sum over n' (two rows in flight; ds_add_f32 resolves collisions)
    {
      const int c = tid & 127;
      const int h0 = tid >> 7;
      const int* secn = sec + (size_t)n * 512;
      for (int np = h0; np < 512; np += 2) {
        int r = secn[np];
        atomicAdd(&pkv[r * 128 + c], S[((size_t)bt * 512 + np) * 128 + c]);
      }
    }
    __syncthreads();

    // f32 -> f16, pad rows 17..31 with zero
    for (int i = tid; i < 32 * 128; i += 256) {
      int r = i >> 7, k = i & 127;
      pa[r * 136 + k] = (r < 17) ? (_Float16)pkv[r * 128 + k] : (_Float16)0.f;
    }
    __syncthreads();

    // KV projection via WMMA: (32x128) x (128x256); wave w owns cols [32w,32w+32)
    {
      const int col0 = wave * 32;
      v8f acc[2][2];
#pragma unroll
      for (int i = 0; i < 2; i++)
#pragma unroll
        for (int j = 0; j < 2; j++)
#pragma unroll
          for (int e = 0; e < 8; e++) acc[i][j][e] = 0.0f;

#pragma unroll
      for (int kc = 0; kc < 128; kc += 32) {
        v16h af[2], bf[2];
#pragma unroll
        for (int mt = 0; mt < 2; mt++)
          af[mt] = frag_a(&pa[(mt * 16 + lr) * 136 + kc + kbA]);
#pragma unroll
        for (int nt = 0; nt < 2; nt++)
          bf[nt] = frag_b(&kvwT[(col0 + nt * 16 + lr) * 136 + kc + kbB]);
#pragma unroll
        for (int mt = 0; mt < 2; mt++)
#pragma unroll
          for (int nt = 0; nt < 2; nt++)
            acc[mt][nt] = __builtin_amdgcn_wmma_f32_16x16x32_f16(
                false, af[mt], false, bf[nt], (short)0, acc[mt][nt], false, false);
      }
#pragma unroll
      for (int mt = 0; mt < 2; mt++)
#pragma unroll
        for (int nt = 0; nt < 2; nt++)
#pragma unroll
          for (int v = 0; v < 8; v++) {
            int r = mt * 16 + rsel + v;
            if (r < 17) kvb[r * 256 + col0 + nt * 16 + lr] = acc[mt][nt][v];
          }
    }
    __syncthreads();

    // Scores (h,r): scale*q.k + bias, mask -> -inf
    if (tid < 68) {
      int h = tid / 17, r = tid % 17;
      float s = 0.f;
#pragma unroll
      for (int e = 0; e < 32; e++) s += qs[h * 32 + e] * kvb[r * 256 + h * 32 + e];
      s = s * 0.17677669529f + bias[h * 17 + r];
      if (mask[(size_t)n * 17 + r]) s = -INFINITY;
      sc[tid] = s;
    }
    __syncthreads();
    if (tid < 4) {
      float m = -INFINITY;
      for (int r = 0; r < 17; r++) m = fmaxf(m, sc[tid * 17 + r]);
      float sum = 0.f;
      for (int r = 0; r < 17; r++) {
        float e = expf(sc[tid * 17 + r] - m);
        pr[tid * 17 + r] = e; sum += e;
      }
      float inv = 1.0f / sum;
      for (int r = 0; r < 17; r++) pr[tid * 17 + r] *= inv;
    }
    __syncthreads();
    if (tid < 128) {
      int h = tid >> 5;
      float o = 0.f;
#pragma unroll
      for (int r = 0; r < 17; r++) o += pr[h * 17 + r] * kvb[r * 256 + 128 + tid];
      ctx[row * 128 + tid] = o;
      ctx16[row * 128 + tid] = (_Float16)o;
    }
    __syncthreads();
  }
}

// ---------------------------------------------------------------------------
// Host orchestration
// ---------------------------------------------------------------------------
extern "C" void kernel_launch(void* const* d_in, const int* in_sizes, int n_in,
                              void* d_out, int out_size, void* d_ws, size_t ws_size,
                              hipStream_t stream) {
  const float* x          = (const float*)d_in[0];
  const float* pos_emb    = (const float*)d_in[1];
  const float* t_qkv_w    = (const float*)d_in[2];
  const float* t_proj_w   = (const float*)d_in[3];
  const float* t_proj_b   = (const float*)d_in[4];
  const float* t_ln_g     = (const float*)d_in[5];
  const float* t_ln_b     = (const float*)d_in[6];
  const float* t_ff_w1    = (const float*)d_in[7];
  const float* t_ff_b1    = (const float*)d_in[8];
  const float* t_ff_w2    = (const float*)d_in[9];
  const float* t_ff_b2    = (const float*)d_in[10];
  const float* assignment = (const float*)d_in[11];
  const unsigned char* s_mask = (const unsigned char*)d_in[12];
  const float* s_q_w      = (const float*)d_in[13];
  const float* s_kv_w     = (const float*)d_in[14];
  const float* s_bias     = (const float*)d_in[15];
  const float* s_proj_w   = (const float*)d_in[16];
  const float* s_proj_b   = (const float*)d_in[17];
  const float* s_ln_g     = (const float*)d_in[18];
  const float* s_ln_b     = (const float*)d_in[19];
  const float* s_ff_w1    = (const float*)d_in[20];
  const float* s_ff_b1    = (const float*)d_in[21];
  const float* s_ff_w2    = (const float*)d_in[22];
  const float* s_ff_b2    = (const float*)d_in[23];
  (void)in_sizes; (void)n_in; (void)ws_size;

  const size_t MT = 49152;  // tokens in both phases
  // fp32 region
  float* h   = (float*)d_ws;            // [MT,128]
  float* h2  = h   + MT * 128;          // [MT,128]
  float* ctx = h2  + MT * 128;          // [MT,128]
  float* aux = ctx + MT * 128;          // [MT,128]
  float* big = aux + MT * 128;          // [MT,512]
  int*   sec = (int*)(big + MT * 512);  // [512,512]
  // f16 region
  _Float16* wh = (_Float16*)(sec + 512 * 512);
  _Float16* h16   = wh; wh += MT * 128;   // mirror of h (temporal) / s (spatial)
  _Float16* ctx16 = wh; wh += MT * 128;
  _Float16* aux16 = wh; wh += MT * 128;
  _Float16* big16 = wh; wh += MT * 512;
  _Float16* wt_qkv  = wh; wh += 2 * 128 * 384;
  _Float16* wt_proj = wh; wh += 2 * 128 * 128;
  _Float16* wt_ff1  = wh; wh += 2 * 128 * 512;
  _Float16* wt_ff2  = wh; wh += 2 * 512 * 128;
  _Float16* wt_q    = wh; wh += 2 * 128 * 128;
  _Float16* wt_kv   = wh; wh += 2 * 128 * 256;
  _Float16* wt_sproj= wh; wh += 2 * 128 * 128;
  _Float16* wt_sff1 = wh; wh += 2 * 128 * 512;
  _Float16* wt_sff2 = wh; wh += 2 * 512 * 128;

  const dim3 blk(256);
  const dim3 gM128(2, 384), gM384(6, 384), gM512(8, 384);

  // Convert + transpose all weights to f16 once.
  auto conv = [&](const float* W, _Float16* WT, int K, int Nn) {
    const int total = K * Nn;
    for (int d = 0; d < 2; d++)
      convertT<<<dim3((total + 255) / 256), blk, 0, stream>>>(
          W + (size_t)d * total, WT + (size_t)d * total, K, Nn);
  };
  conv(t_qkv_w, wt_qkv, 128, 384);
  conv(t_proj_w, wt_proj, 128, 128);
  conv(t_ff_w1, wt_ff1, 128, 512);
  conv(t_ff_w2, wt_ff2, 512, 128);
  conv(s_q_w, wt_q, 128, 128);
  conv(s_kv_w, wt_kv, 128, 256);
  conv(s_proj_w, wt_sproj, 128, 128);
  conv(s_ff_w1, wt_sff1, 128, 512);
  conv(s_ff_w2, wt_sff2, 512, 128);

  // x -> h (+pos_emb), with f16 mirror
  embed_kernel<<<dim3((MT * 128) / 256), blk, 0, stream>>>(x, pos_emb, h, h16);

  // Temporal transformer layers
  for (int d = 0; d < 2; d++) {
    gemm_wmma<0, 0, 0, 0><<<gM384, blk, 0, stream>>>(
        h16, wt_qkv + (size_t)d * 128 * 384, nullptr, nullptr, big, nullptr, (int)MT, 128, 384);
    temporal_attn<<<dim3(16384), blk, 0, stream>>>(big, ctx, ctx16);
    gemm_wmma<1, 0, 1, 0><<<gM128, blk, 0, stream>>>(
        ctx16, wt_proj + (size_t)d * 128 * 128, t_proj_b + d * 128, h, h2, nullptr, (int)MT, 128, 128);
    layernorm<<<dim3(6144), blk, 0, stream>>>(h2, t_ln_g + d * 128, t_ln_b + d * 128, aux, aux16, (int)MT);
    gemm_wmma<1, 1, 0, 1><<<gM512, blk, 0, stream>>>(
        aux16, wt_ff1 + (size_t)d * 128 * 512, t_ff_b1 + d * 512, nullptr, big, big16, (int)MT, 128, 512);
    gemm_wmma<1, 0, 1, 1><<<gM128, blk, 0, stream>>>(
        big16, wt_ff2 + (size_t)d * 512 * 128, t_ff_b2 + d * 128, h2, h, h16, (int)MT, 512, 128);
  }

  // (B*N,T,C) -> (B*T,N,C); h16 becomes the s mirror
  t2s_kernel<<<dim3((MT * 128) / 256), blk, 0, stream>>>(h, h2, h16);
  compute_sec<<<dim3(1024), blk, 0, stream>>>(assignment, sec);

  float* s = h2;
  float* s2 = h;
  const size_t shmem = (size_t)(256 * 136 + 32 * 136) * sizeof(_Float16) +
                       (size_t)(17 * 128 + 17 * 256 + 128 + 68 + 68) * sizeof(float);

  // Spatial transformer layers
  for (int d = 0; d < 2; d++) {
    gemm_wmma<0, 0, 0, 0><<<gM128, blk, 0, stream>>>(
        h16, wt_q + (size_t)d * 128 * 128, nullptr, nullptr, aux, nullptr, (int)MT, 128, 128);
    spatial_attn_fused<<<dim3(128, 96), blk, shmem, stream>>>(
        s, aux, sec, wt_kv + (size_t)d * 128 * 256, s_bias + d * 4 * 17, s_mask, ctx, ctx16);
    gemm_wmma<1, 0, 1, 0><<<gM128, blk, 0, stream>>>(
        ctx16, wt_sproj + (size_t)d * 128 * 128, s_proj_b + d * 128, s, s2, nullptr, (int)MT, 128, 128);
    layernorm<<<dim3(6144), blk, 0, stream>>>(s2, s_ln_g + d * 128, s_ln_b + d * 128, aux, aux16, (int)MT);
    gemm_wmma<1, 1, 0, 1><<<gM512, blk, 0, stream>>>(
        aux16, wt_sff1 + (size_t)d * 128 * 512, s_ff_b1 + d * 512, nullptr, big, big16, (int)MT, 128, 512);
    gemm_wmma<1, 0, 1, 1><<<gM128, blk, 0, stream>>>(
        big16, wt_sff2 + (size_t)d * 512 * 128, s_ff_b2 + d * 128, s2, s, h16, (int)MT, 512, 128);
  }

  // (B*T,N,C) -> (B,C,N,T)
  final_kernel<<<dim3((size_t)out_size / 256), blk, 0, stream>>>(s, (float*)d_out);
}
